// DGN2_70428873720402
// MI455X (gfx1250) — matrix operand order repaired and checked by reading, not compiled, not run on hardware
//
#include <hip/hip_runtime.h>
#include <hip/hip_bf16.h>

// Shapes fixed by the reference harness.
#define T_DIM 4096
#define D_DIM 1024
#define KMAX  16
#define ASTRIDE 1040   // padded LDS row stride in halves (2080B, 16B aligned, bank-skewed)

typedef __attribute__((ext_vector_type(16))) __bf16 v16bf;
typedef __attribute__((ext_vector_type(8)))  float  v8f;

__device__ __forceinline__ float softplusf(float v) {
    return (v > 20.f) ? v : log1pf(expf(v));
}

// float -> bf16, round-to-nearest-even
__device__ __forceinline__ unsigned short f2bf(float f) {
    unsigned int u = __float_as_uint(f);
    unsigned int r = u + 0x7FFFu + ((u >> 16) & 1u);
    return (unsigned short)(r >> 16);
}

// ---------------------------------------------------------------------------
// Kernel 1: per-token stats -> K_t, and normalized rows xn (bf16)
// ---------------------------------------------------------------------------
__global__ __launch_bounds__(256)
void prep_kernel(const float* __restrict__ x,
                 const float* __restrict__ ema_mean,
                 const float* __restrict__ ema_sq,
                 const float* __restrict__ log_sigma_raw,
                 unsigned short* __restrict__ xn,
                 int* __restrict__ Kt, int T)
{
    const int bt  = blockIdx.x;            // flat (b*T + t)
    const int tid = threadIdx.x;
    const float* xr = x + (size_t)bt * D_DIM;

    __shared__ float rsq[256], rab[256];
    float ssq = 0.f, sab = 0.f;
    for (int d = tid; d < D_DIM; d += 256) {
        float xv  = xr[d];
        float m   = ema_mean[d];
        float var = ema_sq[d] - m * m;
        float sd  = sqrtf(fmaxf(var, 1e-6f));
        float z   = (xv - m) / sd;
        ssq += xv * xv;
        sab += fabsf(z);
    }
    rsq[tid] = ssq; rab[tid] = sab;
    __syncthreads();
    for (int s = 128; s > 0; s >>= 1) {
        if (tid < s) { rsq[tid] += rsq[tid + s]; rab[tid] += rab[tid + s]; }
        __syncthreads();
    }
    __shared__ float s_inv;
    if (tid == 0) {
        float norm = sqrtf(rsq[0]);
        s_inv = 1.f / fmaxf(norm, 1e-12f);
        float sigma = softplusf(log_sigma_raw[0]) + 0.01f;
        float surp  = tanhf(sigma * (rab[0] / (float)D_DIM));
        int kmax = (T - 1 < KMAX) ? (T - 1) : KMAX;
        int kt   = (int)rintf(2.0f + 14.0f * surp);   // K_LOW + (K_HIGH-K_LOW)*surp
        kt = kt < 0 ? 0 : (kt > kmax ? kmax : kt);
        Kt[bt] = kt;
    }
    __syncthreads();
    const float inv = s_inv;
    unsigned short* xo = xn + (size_t)bt * D_DIM;
    for (int d = tid; d < D_DIM; d += 256) xo[d] = f2bf(xr[d] * inv);
}

// ---------------------------------------------------------------------------
// Kernel 2: causal cosine-similarity GEMM via V_WMMA_F32_16X16X32_BF16
//           + streaming top-16 selection per row.
// One 128-thread (4-wave) block per (batch, 16-row tile).
// Each wave computes a 16x32 output tile per iteration (2 accumulators,
// shared A fragment) to halve LDS traffic per WMMA.
// ---------------------------------------------------------------------------
__global__ __launch_bounds__(128)
void simtopk_kernel(const unsigned short* __restrict__ xn,
                    const int* __restrict__ Kt,
                    int* __restrict__ topk, int T)
{
    const int rt   = blockIdx.x;          // row tile (16 tokens)
    const int b    = blockIdx.y;
    const int tid  = threadIdx.x;
    const int w    = tid >> 5;
    const int lane = tid & 31;
    const int ln16 = lane & 15;
    const int h    = lane >> 4;
    const int t0   = rt * 16;

    __shared__ __align__(16) unsigned short Asm[16 * ASTRIDE];  // A tile (bf16)
    __shared__ float stage[4][16][32];                          // per-wave 16x32 sim tile
    __shared__ float mv[4][16][16];                             // merge values
    __shared__ int   mi[4][16][16];                             // merge indices

    const unsigned short* xb = xn + (size_t)b * T * D_DIM;

    // cooperative load of the 16-row A tile into LDS
    for (int r = 0; r < 16; ++r) {
        const unsigned int* src = (const unsigned int*)(xb + (size_t)(t0 + r) * D_DIM);
        unsigned int* dst = (unsigned int*)(Asm + r * ASTRIDE);
        for (int i = tid; i < D_DIM / 2; i += 128) dst[i] = src[i];
    }
    __syncthreads();

    // per-lane top-16 list (lanes 0..15 each own one row of the tile)
    float lv[16]; int li[16];
#pragma unroll
    for (int j = 0; j < 16; ++j) { lv[j] = -3.0e38f; li[j] = -1; }
    float vmin = -3.0e38f; int pmin = 0;
    const int trow = t0 + ln16;

    const uint4* arow = (const uint4*)(Asm + ln16 * ASTRIDE);

    // super column tiles of 32 tokens; wave w takes sct = w, w+4, ...
    for (int sct = w; 2 * sct <= rt; sct += 4) {
        const int s0 = sct * 32;
        // second 16-column tile may cross the diagonal / array end: clamp its
        // loads to valid rows; candidates are rejected by the causal guard.
        const int srow1 = (2 * sct + 1 <= rt) ? (s0 + 16 + ln16) : (s0 + ln16);
        const uint4* brow0 = (const uint4*)(xb + (size_t)(s0 + ln16) * D_DIM);
        const uint4* brow1 = (const uint4*)(xb + (size_t)srow1 * D_DIM);
        if (2 * (sct + 4) <= rt) {                     // prefetch next super-tile rows
            __builtin_prefetch(xb + (size_t)(s0 + 128 + ln16) * D_DIM, 0, 1);
            __builtin_prefetch(xb + (size_t)(s0 + 144 + ln16) * D_DIM, 0, 1);
        }

        v8f acc0 = {};
        v8f acc1 = {};
#pragma unroll 4
        for (int kk = 0; kk < D_DIM / 32; ++kk) {
            union { v16bf v; uint4 u[2]; } af, bf0, bf1;
            // A (16x32, MxK): lane holds row M=ln16, K in [h*16, h*16+16) contiguous
            af.u[0] = arow[kk * 4 + 2 * h];
            af.u[1] = arow[kk * 4 + 2 * h + 1];
            // B (32x16, KxN): lane holds col N=ln16, K in [h*16, h*16+16) contiguous
            bf0.u[0] = brow0[kk * 4 + 2 * h];
            bf0.u[1] = brow0[kk * 4 + 2 * h + 1];
            bf1.u[0] = brow1[kk * 4 + 2 * h];
            bf1.u[1] = brow1[kk * 4 + 2 * h + 1];
            acc0 = __builtin_amdgcn_wmma_f32_16x16x32_bf16(
                       false, af.v, false, bf0.v, (short)0, acc0, false, false);
            acc1 = __builtin_amdgcn_wmma_f32_16x16x32_bf16(
                       false, af.v, false, bf1.v, (short)0, acc1, false, false);
        }

        // stage the 16x32 f32 tile: C layout -> M = vv + 8*h, N = ln16 (+16)
#pragma unroll
        for (int vv = 0; vv < 8; ++vv) {
            stage[w][vv + 8 * h][ln16]      = acc0[vv];
            stage[w][vv + 8 * h][ln16 + 16] = acc1[vv];
        }
        asm volatile("s_wait_dscnt 0" ::: "memory");   // intra-wave LDS ordering

        if (lane < 16) {                               // lane r updates row r's top-16
#pragma unroll
            for (int n = 0; n < 32; ++n) {
                float val = stage[w][lane][n];
                int   s   = s0 + n;                    // ==(2sct)*16+n / (2sct+1)*16+n'
                if (s < trow && val > vmin) {          // strict causal + replace-min
#pragma unroll
                    for (int j = 0; j < 16; ++j)
                        if (j == pmin) { lv[j] = val; li[j] = s; }
                    vmin = lv[0]; pmin = 0;
#pragma unroll
                    for (int j = 1; j < 16; ++j)
                        if (lv[j] < vmin) { vmin = lv[j]; pmin = j; }
                }
            }
        }
        asm volatile("s_wait_dscnt 0" ::: "memory");
    }

    // publish per-wave lists and merge
    if (lane < 16) {
#pragma unroll
        for (int j = 0; j < 16; ++j) { mv[w][lane][j] = lv[j]; mi[w][lane][j] = li[j]; }
    }
    __syncthreads();

    if (tid < 16) {
        const int r = tid;
        float fv[16]; int fi[16];
#pragma unroll
        for (int j = 0; j < 16; ++j) { fv[j] = -3.0e38f; fi[j] = -1; }
        float vm = -3.0e38f; int pm = 0;
        for (int ww = 0; ww < 4; ++ww) {
#pragma unroll
            for (int j = 0; j < 16; ++j) {
                float val = mv[ww][r][j]; int s = mi[ww][r][j];
                if (s >= 0 && val > vm) {
#pragma unroll
                    for (int q = 0; q < 16; ++q)
                        if (q == pm) { fv[q] = val; fi[q] = s; }
                    vm = fv[0]; pm = 0;
#pragma unroll
                    for (int q = 1; q < 16; ++q)
                        if (fv[q] < vm) { vm = fv[q]; pm = q; }
                }
            }
        }
        // sort desc by value, tie-break by smaller index (stable-argsort-like)
#pragma unroll
        for (int i2 = 0; i2 < 15; ++i2) {
#pragma unroll
            for (int j2 = i2 + 1; j2 < 16; ++j2) {
                bool sw = (fv[j2] > fv[i2]) ||
                          (fv[j2] == fv[i2] &&
                           ((unsigned)fi[j2] < (unsigned)fi[i2]));
                if (sw) {
                    float tv = fv[i2]; fv[i2] = fv[j2]; fv[j2] = tv;
                    int   ti = fi[i2]; fi[i2] = fi[j2]; fi[j2] = ti;
                }
            }
        }
        const int t  = t0 + r;
        const int kt = Kt[b * T + t];
        int* outp = topk + (size_t)(b * T + t) * 16;
#pragma unroll
        for (int j = 0; j < 16; ++j) outp[j] = (j < kt) ? fi[j] : -1;
    }
}

// ---------------------------------------------------------------------------
// Kernel 3: sparse neighbor mean + blend + exact GELU epilogue
// ---------------------------------------------------------------------------
__global__ __launch_bounds__(256)
void aggregate_kernel(const float* __restrict__ x,
                      const float* __restrict__ gain,
                      const float* __restrict__ bias,
                      const float* __restrict__ log_mix,
                      const float* __restrict__ log_scale,
                      const int* __restrict__ topk,
                      float* __restrict__ out, int T)
{
    const int bt  = blockIdx.x;
    const int tid = threadIdx.x;
    __shared__ int sIdx[16];
    __shared__ int sCnt;
    if (tid < 16) sIdx[tid] = topk[(size_t)bt * 16 + tid];
    __syncthreads();
    if (tid == 0) {
        int c = 0;
#pragma unroll
        for (int j = 0; j < 16; ++j) c += (sIdx[j] >= 0);
        sCnt = c;
    }
    __syncthreads();

    const float deg     = (float)(sCnt > 1 ? sCnt : 1);
    const float inv_deg = 1.f / deg;
    const float mix     = 1.f / (1.f + expf(-log_mix[0]));
    const float scale   = softplusf(log_scale[0]) + 0.01f;
    const int   b       = bt / T;
    const float* xbase  = x + (size_t)b * T * D_DIM;
    const float* xr     = x + (size_t)bt * D_DIM;

    for (int d = tid; d < D_DIM; d += 256) {
        float acc = 0.f;
#pragma unroll
        for (int j = 0; j < 16; ++j) {
            int s = sIdx[j];
            if (s >= 0) acc += xbase[(size_t)s * D_DIM + d];
        }
        float xv      = xr[d];
        float msg     = acc * inv_deg;
        float blended = mix * xv + (1.f - mix) * msg;
        float u       = blended * gain[d] + bias[d];
        float g       = 0.5f * u * (1.f + erff(u * 0.70710678118654752f));
        out[(size_t)bt * D_DIM + d] = g * scale;
    }
}

// ---------------------------------------------------------------------------
extern "C" void kernel_launch(void* const* d_in, const int* in_sizes, int n_in,
                              void* d_out, int out_size, void* d_ws, size_t ws_size,
                              hipStream_t stream)
{
    const float* x          = (const float*)d_in[0];
    const float* gain       = (const float*)d_in[1];
    const float* bias       = (const float*)d_in[2];
    const float* log_mix    = (const float*)d_in[3];
    const float* log_scale  = (const float*)d_in[4];
    const float* log_sigma  = (const float*)d_in[5];
    // d_in[6] = logit_decay (unused by the reference output)
    const float* ema_mean   = (const float*)d_in[7];
    const float* ema_sq     = (const float*)d_in[8];
    float* out = (float*)d_out;

    const int D  = in_sizes[1];                 // 1024
    const int T  = T_DIM;                       // 4096 (fixed by harness)
    const int B  = in_sizes[0] / (T * D);       // 2
    const int BT = B * T;

    // workspace layout
    char* ws = (char*)d_ws;
    size_t xnBytes = (size_t)BT * D * sizeof(unsigned short);   // 16 MB
    unsigned short* xn = (unsigned short*)ws;
    int* Kt   = (int*)(ws + xnBytes);
    int* topk = (int*)(ws + xnBytes + (size_t)BT * sizeof(int));

    prep_kernel<<<BT, 256, 0, stream>>>(x, ema_mean, ema_sq, log_sigma, xn, Kt, T);

    dim3 gB(T / 16, B);
    simtopk_kernel<<<gB, 128, 0, stream>>>(xn, Kt, topk, T);

    aggregate_kernel<<<BT, 256, 0, stream>>>(x, gain, bias, log_mix, log_scale,
                                             topk, out, T);
}
